// GraphNetLayer_63453846831576
// MI455X (gfx1250) — matrix operand order repaired and checked by reading, not compiled, not run on hardware
//
#include <hip/hip_runtime.h>

#define NN    2048
#define TT    512
#define NIN   16
#define NOUT  32
#define SLOPE 0.01f
#define EPS   1e-5f
#define WAVES 8
#define NTHREADS 256
#define TILES (NN * (TT / 16))
#define SCR_STRIDE 40   // halves per row: 32 used + 8 pad -> 80B row, 16B-aligned B128 access

typedef __attribute__((ext_vector_type(16))) _Float16 v16h;
typedef __attribute__((ext_vector_type(8)))  _Float16 v8h;
typedef __attribute__((ext_vector_type(2)))  _Float16 v2h;
typedef __attribute__((ext_vector_type(8)))  float    v8f;

__device__ __forceinline__ v8f wmma(v16h a, v16h b, v8f c) {
  return __builtin_amdgcn_wmma_f32_16x16x32_f16(false, a, false, b, (short)0, c, false, false);
}

// ---- A fragment: 16x32 f16; lane m=l%16, h=l/16: elems 0-7 = pos[8h..8h+7], 8-15 = pos[16+8h..23+8h]
__device__ __forceinline__ v16h load_A(const _Float16* scr, int lane) {
  int m = lane & 15, h = lane >> 4;
  const _Float16* p = scr + m * SCR_STRIDE + 8 * h;
  v8h lo = *(const v8h*)p;          // ds_load_b128
  v8h hi = *(const v8h*)(p + 16);   // ds_load_b128
  return __builtin_shufflevector(lo, hi, 0,1,2,3,4,5,6,7,8,9,10,11,12,13,14,15);
}

// ---- 6 B fragments (3 layers x 2 ntiles) held in VGPRs
struct BReg { v16h f[6]; };
__device__ __forceinline__ BReg load_Bregs(const _Float16* Bf, int lane) {
  BReg r;
#pragma unroll
  for (int i = 0; i < 6; ++i) r.f[i] = *(const v16h*)(Bf + (i * 32 + lane) * 16);
  return r;
}

// ---- D -> bias + LeakyReLU -> packed f16 pair at positions (2c, 2c+1) = units (c, c+16)
__device__ __forceinline__ void act_store_pk(_Float16* scr, const v8f& c0, const v8f& c1,
                                             float b0, float b1, int lane) {
  int h = lane >> 4, c = lane & 15;
#pragma unroll
  for (int v = 0; v < 8; ++v) {
    int row = v + 8 * h;
    float x0 = c0[v] + b0; x0 = fmaxf(x0, SLOPE * x0);
    float x1 = c1[v] + b1; x1 = fmaxf(x1, SLOPE * x1);
    v2h pk; pk[0] = (_Float16)x0; pk[1] = (_Float16)x1;
    *(v2h*)(scr + row * SCR_STRIDE + 2 * c) = pk;     // single ds_store_b32
  }
}

// ---- 3-layer MLP on a 16-row tile; pre-BN h returned in D layout (d0: cols 0-15, d1: 16-31)
__device__ __forceinline__ void mlp3(_Float16* scr, const BReg& B,
                                     float b1l, float b1h, float b2l, float b2h,
                                     float b3l, float b3h, int lane,
                                     v8f& d0, v8f& d1) {
  v8f z = {0.f,0.f,0.f,0.f,0.f,0.f,0.f,0.f};
  v16h a = load_A(scr, lane);
  v8f c0 = wmma(a, B.f[0], z);
  v8f c1 = wmma(a, B.f[1], z);
  act_store_pk(scr, c0, c1, b1l, b1h, lane);
  a  = load_A(scr, lane);
  c0 = wmma(a, B.f[2], z);
  c1 = wmma(a, B.f[3], z);
  act_store_pk(scr, c0, c1, b2l, b2h, lane);
  a  = load_A(scr, lane);
  c0 = wmma(a, B.f[4], z);
  c1 = wmma(a, B.f[5], z);
#pragma unroll
  for (int v = 0; v < 8; ++v) { d0[v] = c0[v] + b3l; d1[v] = c1[v] + b3h; }
}

// ---- stage 16x32 edge input tile: lane = (row m, half); one 64B x-row per lane
__device__ __forceinline__ void stage_input(_Float16* scr, const float* __restrict__ x,
                                            int nbase, int t0, int mode, int lane) {
  int m = lane & 15, half = lane >> 4;
  int t = t0 + m;
  int tn = (mode == 1) ? ((t > 0) ? t - 1 : 0)
         : (mode == 2) ? ((t < TT - 1) ? t + 1 : TT - 1) : t;
  int ts = half ? tn : t;
  const float4* p = (const float4*)(x + nbase + ts * NIN);
  float4 f0 = p[0], f1 = p[1], f2 = p[2], f3 = p[3];  // 4x global_load_b128
  v8h lo, hi;
  lo[0]=(_Float16)f0.x; lo[1]=(_Float16)f0.y; lo[2]=(_Float16)f0.z; lo[3]=(_Float16)f0.w;
  lo[4]=(_Float16)f1.x; lo[5]=(_Float16)f1.y; lo[6]=(_Float16)f1.z; lo[7]=(_Float16)f1.w;
  hi[0]=(_Float16)f2.x; hi[1]=(_Float16)f2.y; hi[2]=(_Float16)f2.z; hi[3]=(_Float16)f2.w;
  hi[4]=(_Float16)f3.x; hi[5]=(_Float16)f3.y; hi[6]=(_Float16)f3.z; hi[7]=(_Float16)f3.w;
  v8h* dst = (v8h*)(scr + m * SCR_STRIDE + 16 * half);
  dst[0] = lo; dst[1] = hi;                            // 2x ds_store_b128
}

// ---- build swizzled B fragments. Position p maps to weight row p (natural) or
//      u(p) = (p>>1) + 16*(p&1) (packed-pair layout produced by act_store_pk).
//      Layers 2/3 always permuted; layer 1 permuted iff its input comes packed (node MLP).
__device__ __forceinline__ void stage_weights(_Float16* Bf, const float* W1,
                                              const float* W2, const float* W3,
                                              bool perm1, int tid) {
  for (int idx = tid; idx < 3 * 2 * 32 * 16; idx += NTHREADS) {
    int e  = idx & 15;
    int L  = (idx >> 4) & 31;
    int nt = (idx >> 9) & 1;
    int l  = idx >> 10;
    int p  = (e & 7) + 8 * (L >> 4) + 16 * (e >> 3);
    bool perm = (l > 0) || perm1;
    int K  = perm ? ((p >> 1) + 16 * (p & 1)) : p;
    int col = (L & 15) + 16 * nt;
    const float* W = (l == 0) ? W1 : ((l == 1) ? W2 : W3);
    Bf[idx] = (_Float16)W[K * 32 + col];
  }
}

// ======================= Pass 0: zero stat accumulators =======================
__global__ void gn_init_ws(double* dsum) {
  int i = blockIdx.x * blockDim.x + threadIdx.x;
  if (i < 256) dsum[i] = 0.0;
}

// ======================= Pass 1: edge MLP batch statistics =======================
__global__ __launch_bounds__(NTHREADS)
void gn_edge_stats(const float* __restrict__ x,
                   const float* __restrict__ W1, const float* __restrict__ b1,
                   const float* __restrict__ W2, const float* __restrict__ b2,
                   const float* __restrict__ W3, const float* __restrict__ b3,
                   double* __restrict__ dsum) {
  __shared__ __attribute__((aligned(32))) _Float16 s_Bf[3 * 2 * 32 * 16];
  __shared__ __attribute__((aligned(16))) _Float16 s_scr[WAVES][16 * SCR_STRIDE];
  __shared__ float s_b1[3][32], s_b2[32], s_b3[32];
  __shared__ double s_stat[3 * 64];

  int tid = threadIdx.x, lane = tid & 31, wid = tid >> 5;
  stage_weights(s_Bf, W1, W2, W3, false, tid);
  for (int c = tid; c < 32; c += NTHREADS) {
    float bb = b1[c], w32 = W1[32 * 32 + c];          // fold dir * W1[32,:] into bias
    s_b1[0][c] = bb; s_b1[1][c] = bb + w32; s_b1[2][c] = bb - w32;
    s_b2[c] = b2[c]; s_b3[c] = b3[c];
  }
  for (int i = tid; i < 3 * 64; i += NTHREADS) s_stat[i] = 0.0;
  __syncthreads();

  _Float16* scr = s_scr[wid];
  int h = lane >> 4, c = lane & 15;
  // loop-invariant operands hoisted into registers
  BReg B = load_Bregs(s_Bf, lane);
  float b1l0 = s_b1[0][c], b1h0 = s_b1[0][c + 16];
  float b1l1 = s_b1[1][c], b1h1 = s_b1[1][c + 16];
  float b1l2 = s_b1[2][c], b1h2 = s_b1[2][c + 16];
  float b2l = s_b2[c], b2h = s_b2[c + 16];
  float b3l = s_b3[c], b3h = s_b3[c + 16];

  int wave = blockIdx.x * WAVES + wid;
  int wstride = gridDim.x * WAVES;
  for (int tile = wave; tile < TILES; tile += wstride) {
    int n = tile >> 5;
    int t0 = (tile & 31) << 4;
    int nbase = n * (TT * NIN);
    auto do_mode = [&](int mode, float bl, float bh) {
      stage_input(scr, x, nbase, t0, mode, lane);
      v8f d0, d1;
      mlp3(scr, B, bl, bh, b2l, b2h, b3l, b3h, lane, d0, d1);
      float s0 = 0.f, q0 = 0.f, s1 = 0.f, q1 = 0.f;
#pragma unroll
      for (int v = 0; v < 8; ++v) {
        int t = t0 + v + 8 * h;
        bool valid = (mode == 0) || (mode == 1 && t >= 1) || (mode == 2 && t <= TT - 2);
        if (valid) { s0 += d0[v]; q0 += d0[v] * d0[v]; s1 += d1[v]; q1 += d1[v] * d1[v]; }
      }
      atomicAdd(&s_stat[mode * 64 + c],      (double)s0);
      atomicAdd(&s_stat[mode * 64 + 32 + c], (double)q0);
      atomicAdd(&s_stat[mode * 64 + 16 + c], (double)s1);
      atomicAdd(&s_stat[mode * 64 + 48 + c], (double)q1);
    };
    do_mode(0, b1l0, b1h0);
    do_mode(1, b1l1, b1h1);
    do_mode(2, b1l2, b1h2);
  }
  __syncthreads();
  for (int i = tid; i < 3 * 64; i += NTHREADS) atomicAdd(&dsum[i], s_stat[i]);
}

// ======================= Pass 2: edge BN scale/shift =======================
__global__ void gn_finalize_edge(const double* __restrict__ dsum, float* __restrict__ fsc,
                                 const float* __restrict__ gamma, const float* __restrict__ beta) {
  int i = threadIdx.x;
  if (i < 96) {
    int call = i >> 5, c = i & 31;
    double cnt = (call == 0) ? (double)NN * TT : (double)NN * (TT - 1);
    double mu  = dsum[call * 64 + c] / cnt;
    double var = dsum[call * 64 + 32 + c] / cnt - mu * mu;
    float sc = gamma[c] * rsqrtf((float)var + EPS);
    fsc[call * 64 + c]      = sc;
    fsc[call * 64 + 32 + c] = beta[c] - (float)mu * sc;
  }
}

// ============ Pass 3: recompute edges, normalize, aggregate, node MLP + stats ============
__global__ __launch_bounds__(NTHREADS)
void gn_node_pass(const float* __restrict__ x,
                  const float* eW1, const float* eb1, const float* eW2, const float* eb2,
                  const float* eW3, const float* eb3,
                  const float* nW1, const float* nb1, const float* nW2, const float* nb2,
                  const float* nW3, const float* nb3,
                  const float* __restrict__ fsc, double* __restrict__ dsum,
                  float* __restrict__ out) {
  __shared__ __attribute__((aligned(32))) _Float16 s_Bfe[3 * 2 * 32 * 16];
  __shared__ __attribute__((aligned(32))) _Float16 s_Bfn[3 * 2 * 32 * 16];
  __shared__ __attribute__((aligned(16))) _Float16 s_scr[WAVES][16 * SCR_STRIDE];
  __shared__ float s_b1[3][32], s_b2e[32], s_b3e[32];
  __shared__ float s_b1n[32], s_b2n[32], s_b3n[32];
  __shared__ float s_sc[3 * 64];
  __shared__ double s_stat[64];

  int tid = threadIdx.x, lane = tid & 31, wid = tid >> 5;
  stage_weights(s_Bfe, eW1, eW2, eW3, false, tid);
  stage_weights(s_Bfn, nW1, nW2, nW3, true, tid);   // node layer-1 input arrives packed
  for (int c = tid; c < 32; c += NTHREADS) {
    float bb = eb1[c], w32 = eW1[32 * 32 + c];
    s_b1[0][c] = bb; s_b1[1][c] = bb + w32; s_b1[2][c] = bb - w32;
    s_b2e[c] = eb2[c]; s_b3e[c] = eb3[c];
    s_b1n[c] = nb1[c]; s_b2n[c] = nb2[c]; s_b3n[c] = nb3[c];
  }
  for (int i = tid; i < 3 * 64; i += NTHREADS) s_sc[i] = fsc[i];
  for (int i = tid; i < 64; i += NTHREADS) s_stat[i] = 0.0;
  __syncthreads();

  _Float16* scr = s_scr[wid];
  int h = lane >> 4, c = lane & 15;
  BReg Be = load_Bregs(s_Bfe, lane);
  float b1l0 = s_b1[0][c], b1h0 = s_b1[0][c + 16];
  float b1l1 = s_b1[1][c], b1h1 = s_b1[1][c + 16];
  float b1l2 = s_b1[2][c], b1h2 = s_b1[2][c + 16];
  float b2le = s_b2e[c], b2he = s_b2e[c + 16];
  float b3le = s_b3e[c], b3he = s_b3e[c + 16];
  float b1ln = s_b1n[c], b1hn = s_b1n[c + 16];
  float b2ln = s_b2n[c], b2hn = s_b2n[c + 16];
  float b3ln = s_b3n[c], b3hn = s_b3n[c + 16];
  float sci0 = s_sc[c],       shi0 = s_sc[32 + c];
  float sci1 = s_sc[16 + c],  shi1 = s_sc[48 + c];
  float scf0 = s_sc[64 + c],  shf0 = s_sc[96 + c];
  float scf1 = s_sc[80 + c],  shf1 = s_sc[112 + c];
  float scb0 = s_sc[128 + c], shb0 = s_sc[160 + c];
  float scb1 = s_sc[144 + c], shb1 = s_sc[176 + c];

  int wave = blockIdx.x * WAVES + wid;
  int wstride = gridDim.x * WAVES;
  for (int tile = wave; tile < TILES; tile += wstride) {
    int n = tile >> 5;
    int t0 = (tile & 31) << 4;
    int nbase = n * (TT * NIN);
    v8f di0, di1, df0, df1, db0, db1;
    stage_input(scr, x, nbase, t0, 0, lane);
    mlp3(scr, Be, b1l0, b1h0, b2le, b2he, b3le, b3he, lane, di0, di1);
    stage_input(scr, x, nbase, t0, 1, lane);
    mlp3(scr, Be, b1l1, b1h1, b2le, b2he, b3le, b3he, lane, df0, df1);
    stage_input(scr, x, nbase, t0, 2, lane);
    mlp3(scr, Be, b1l2, b1h2, b2le, b2he, b3le, b3he, lane, db0, db1);
    // normalize + aggregate (zero-padded shifts) -> packed node input in scratch
#pragma unroll
    for (int v = 0; v < 8; ++v) {
      int row = v + 8 * h, t = t0 + row;
      float a0 = di0[v] * sci0 + shi0;
      float a1 = di1[v] * sci1 + shi1;
      if (t >= 1)      { a0 += df0[v] * scf0 + shf0; a1 += df1[v] * scf1 + shf1; }
      if (t <= TT - 2) { a0 += db0[v] * scb0 + shb0; a1 += db1[v] * scb1 + shb1; }
      v2h pk; pk[0] = (_Float16)a0; pk[1] = (_Float16)a1;
      *(v2h*)(scr + row * SCR_STRIDE + 2 * c) = pk;
    }
    BReg Bn = load_Bregs(s_Bfn, lane);
    v8f dn0, dn1;
    mlp3(scr, Bn, b1ln, b1hn, b2ln, b2hn, b3ln, b3hn, lane, dn0, dn1);
    float s0 = 0.f, q0 = 0.f, s1 = 0.f, q1 = 0.f;
#pragma unroll
    for (int v = 0; v < 8; ++v) {
      int t = t0 + v + 8 * h;
      int o = (n * TT + t) * NOUT;
      out[o + c]      = dn0[v];     // pre-BN node activations staged in d_out
      out[o + c + 16] = dn1[v];
      s0 += dn0[v]; q0 += dn0[v] * dn0[v]; s1 += dn1[v]; q1 += dn1[v] * dn1[v];
    }
    atomicAdd(&s_stat[c],      (double)s0);
    atomicAdd(&s_stat[32 + c], (double)q0);
    atomicAdd(&s_stat[16 + c], (double)s1);
    atomicAdd(&s_stat[48 + c], (double)q1);
  }
  __syncthreads();
  for (int i = tid; i < 64; i += NTHREADS) atomicAdd(&dsum[192 + i], s_stat[i]);
}

// ======================= Pass 4: node BN scale/shift =======================
__global__ void gn_finalize_node(const double* __restrict__ dsum, float* __restrict__ fsc,
                                 const float* __restrict__ gamma, const float* __restrict__ beta) {
  int c = threadIdx.x;
  if (c < 32) {
    double cnt = (double)NN * TT;
    double mu  = dsum[192 + c] / cnt;
    double var = dsum[192 + 32 + c] / cnt - mu * mu;
    float sc = gamma[c] * rsqrtf((float)var + EPS);
    fsc[192 + c]      = sc;
    fsc[192 + 32 + c] = beta[c] - (float)mu * sc;
  }
}

// ======================= Pass 5: in-place BN apply on d_out (float4) =======================
__global__ __launch_bounds__(NTHREADS)
void gn_bn_apply(float* __restrict__ out, const float* __restrict__ fsc) {
  __shared__ float s_sc[32], s_sh[32];
  if (threadIdx.x < 32) {
    s_sc[threadIdx.x] = fsc[192 + threadIdx.x];
    s_sh[threadIdx.x] = fsc[224 + threadIdx.x];
  }
  __syncthreads();
  const int total4 = NN * TT * (NOUT / 4);
  for (int i = blockIdx.x * NTHREADS + threadIdx.x; i < total4; i += gridDim.x * NTHREADS) {
    float4 v = ((const float4*)out)[i];
    int col = (i & 7) * 4;
    v.x = v.x * s_sc[col]     + s_sh[col];
    v.y = v.y * s_sc[col + 1] + s_sh[col + 1];
    v.z = v.z * s_sc[col + 2] + s_sh[col + 2];
    v.w = v.w * s_sc[col + 3] + s_sh[col + 3];
    ((float4*)out)[i] = v;
  }
}

extern "C" void kernel_launch(void* const* d_in, const int* in_sizes, int n_in,
                              void* d_out, int out_size, void* d_ws, size_t ws_size,
                              hipStream_t stream) {
  const float* x   = (const float*)d_in[0];
  const float* eW1 = (const float*)d_in[1];
  const float* eb1 = (const float*)d_in[2];
  const float* eW2 = (const float*)d_in[3];
  const float* eb2 = (const float*)d_in[4];
  const float* eW3 = (const float*)d_in[5];
  const float* eb3 = (const float*)d_in[6];
  const float* egm = (const float*)d_in[7];
  const float* ebt = (const float*)d_in[8];
  const float* nW1 = (const float*)d_in[9];
  const float* nb1 = (const float*)d_in[10];
  const float* nW2 = (const float*)d_in[11];
  const float* nb2 = (const float*)d_in[12];
  const float* nW3 = (const float*)d_in[13];
  const float* nb3 = (const float*)d_in[14];
  const float* ngm = (const float*)d_in[15];
  const float* nbt = (const float*)d_in[16];
  float* out = (float*)d_out;
  double* dsum = (double*)d_ws;                       // 256 doubles: sums/sumsqs (i,f,b,node)
  float*  fsc  = (float*)((char*)d_ws + 2048);        // 256 floats: scale/shift (i,f,b,node)

  gn_init_ws<<<dim3(1), dim3(256), 0, stream>>>(dsum);
  gn_edge_stats<<<dim3(2048), dim3(NTHREADS), 0, stream>>>(x, eW1, eb1, eW2, eb2, eW3, eb3, dsum);
  gn_finalize_edge<<<dim3(1), dim3(128), 0, stream>>>(dsum, fsc, egm, ebt);
  gn_node_pass<<<dim3(2048), dim3(NTHREADS), 0, stream>>>(x, eW1, eb1, eW2, eb2, eW3, eb3,
                                                          nW1, nb1, nW2, nb2, nW3, nb3,
                                                          fsc, dsum, out);
  gn_finalize_node<<<dim3(1), dim3(64), 0, stream>>>(dsum, fsc, ngm, nbt);
  gn_bn_apply<<<dim3(2048), dim3(NTHREADS), 0, stream>>>(out, fsc);
}
